// MixtureOfExperts_34059090657428
// MI455X (gfx1250) — matrix operand order repaired and checked by reading, not compiled, not run on hardware
//
#include <hip/hip_runtime.h>
#include <hip/hip_bf16.h>

#define B_TOK 16384
#define DIM   2048
#define NEXP  8

// fallback tile
#define BM 32
#define BN 256
#define BK 32
// packed-path tile
#define PBM 64
#define PBN 256
#define NTILES (DIM / 16)   // 128 n-tiles per expert
#define KBLKS  (DIM / 32)   // 64 k-blocks

typedef __attribute__((ext_vector_type(16))) __bf16 v16bf;
typedef __attribute__((ext_vector_type(8)))  __bf16 bf16x8;
typedef __attribute__((ext_vector_type(8)))  float  v8f;

static __device__ __forceinline__ __bf16 f2bf(float f) {
    // round-to-nearest-even f32 -> bf16 (inputs finite)
    unsigned u = __builtin_bit_cast(unsigned, f);
    unsigned r = u + 0x7FFFu + ((u >> 16) & 1u);
    unsigned short h = (unsigned short)(r >> 16);
    return __builtin_bit_cast(__bf16, h);
}

// ---------------------------------------------------------------- zero init
__global__ void moe_zero(float* __restrict__ out, int* __restrict__ cnt, long n4) {
    long i = (long)blockIdx.x * blockDim.x + threadIdx.x;
    long stride = (long)gridDim.x * blockDim.x;
    float4* o4 = (float4*)out;
    for (long j = i; j < n4; j += stride) o4[j] = make_float4(0.f, 0.f, 0.f, 0.f);
    if (blockIdx.x == 0 && threadIdx.x < NEXP) cnt[threadIdx.x] = 0;
}

// ---------------------------------------------------------------- gate + route
__global__ void __launch_bounds__(256) moe_gate(
    const float* __restrict__ x, const float* __restrict__ gw,
    const float* __restrict__ gb, int* __restrict__ cnt,
    int* __restrict__ tok, float* __restrict__ wgt)
{
    const int wid  = threadIdx.x >> 5;
    const int lane = threadIdx.x & 31;
    const int t    = blockIdx.x * 8 + wid;

    const float* xr = x + (size_t)t * DIM;
    float acc[NEXP];
#pragma unroll
    for (int e = 0; e < NEXP; ++e) acc[e] = 0.f;

    for (int d = lane; d < DIM; d += 32) {
        float xv = xr[d];
        const float4 g0 = ((const float4*)(gw + (size_t)d * NEXP))[0];
        const float4 g1 = ((const float4*)(gw + (size_t)d * NEXP))[1];
        acc[0] = fmaf(xv, g0.x, acc[0]);
        acc[1] = fmaf(xv, g0.y, acc[1]);
        acc[2] = fmaf(xv, g0.z, acc[2]);
        acc[3] = fmaf(xv, g0.w, acc[3]);
        acc[4] = fmaf(xv, g1.x, acc[4]);
        acc[5] = fmaf(xv, g1.y, acc[5]);
        acc[6] = fmaf(xv, g1.z, acc[6]);
        acc[7] = fmaf(xv, g1.w, acc[7]);
    }
#pragma unroll
    for (int off = 16; off > 0; off >>= 1) {
#pragma unroll
        for (int e = 0; e < NEXP; ++e)
            acc[e] += __shfl_xor(acc[e], off, 32);
    }

    if (lane == 0) {
        float l[NEXP];
#pragma unroll
        for (int e = 0; e < NEXP; ++e) l[e] = acc[e] + gb[e];

        int b0 = 0; float m0 = l[0];
#pragma unroll
        for (int e = 1; e < NEXP; ++e) if (l[e] > m0) { m0 = l[e]; b0 = e; }
        int b1 = -1; float m1 = -3.4e38f;
#pragma unroll
        for (int e = 0; e < NEXP; ++e) if (e != b0 && l[e] > m1) { m1 = l[e]; b1 = e; }

        float e2  = __expf(m1 - m0);
        float inv = 1.f / (1.f + e2);
        float w0  = inv, w1 = e2 * inv;

        int p0 = atomicAdd(&cnt[b0], 1);
        tok[b0 * B_TOK + p0] = t;  wgt[b0 * B_TOK + p0] = w0;
        int p1 = atomicAdd(&cnt[b1], 1);
        tok[b1 * B_TOK + p1] = t;  wgt[b1 * B_TOK + p1] = w1;
    }
}

// ---------------------------------------------------------------- x pre-convert
// straight f32 -> bf16, row-major (B, D); rows keep A-fragment byte layout
__global__ void __launch_bounds__(256) moe_pack_x(
    const float* __restrict__ x, __bf16* __restrict__ xb, long n8)
{
    long i = (long)blockIdx.x * blockDim.x + threadIdx.x;
    long stride = (long)gridDim.x * blockDim.x;
    for (long j = i; j < n8; j += stride) {
        float4 f0 = ((const float4*)x)[2 * j];
        float4 f1 = ((const float4*)x)[2 * j + 1];
        bf16x8 o;
        o[0] = f2bf(f0.x); o[1] = f2bf(f0.y); o[2] = f2bf(f0.z); o[3] = f2bf(f0.w);
        o[4] = f2bf(f1.x); o[5] = f2bf(f1.y); o[6] = f2bf(f1.z); o[7] = f2bf(f1.w);
        ((bf16x8*)xb)[j] = o;
    }
}

// ---------------------------------------------------------------- W pre-pack
// expert_w (f32, [e][k][n]) -> bf16 tiles in WMMA B-fragment order:
// tile (e, nt, kb): 512 bf16; lane l's 16 elems at l*16 are
// W[kb*32 + 8*(l>>4) + v + (v<8?0:8)][nt*16 + (l&15)], v=0..15.
// Tile order ((e*NTILES + nt)*KBLKS + kb) -> consecutive kb contiguous.
__global__ void __launch_bounds__(256) moe_pack_w(
    const float* __restrict__ ew, __bf16* __restrict__ wp)
{
    const int nchunk = blockIdx.x;     // 0..7
    const int kb     = blockIdx.y;     // 0..63
    const int e      = blockIdx.z;     // 0..7
    const int tid    = threadIdx.x;

    __shared__ __bf16 T[256][32];      // T[n_local][k_local], 16 KB

    const float* ewp = ew + (size_t)e * DIM * DIM;
#pragma unroll
    for (int i = 0; i < 8; ++i) {
        int idx = tid + 256 * i;
        int kk  = idx >> 6;
        int nn  = (idx & 63) * 4;
        float4 wv = *(const float4*)(ewp + (size_t)(kb * 32 + kk) * DIM +
                                     nchunk * 256 + nn);
        T[nn + 0][kk] = f2bf(wv.x);
        T[nn + 1][kk] = f2bf(wv.y);
        T[nn + 2][kk] = f2bf(wv.z);
        T[nn + 3][kk] = f2bf(wv.w);
    }
    __syncthreads();

    const int j  = tid >> 4;           // local n-tile 0..15
    const int ln = tid & 15;
    const int nt = nchunk * 16 + j;
    __bf16* tb = wp + (((size_t)e * NTILES + nt) * KBLKS + kb) * 512;

    const __bf16* row = &T[j * 16 + ln][0];
    ((bf16x8*)(tb + ln * 16))[0]        = ((const bf16x8*)row)[0];
    ((bf16x8*)(tb + ln * 16))[1]        = ((const bf16x8*)row)[2];
    ((bf16x8*)(tb + (ln + 16) * 16))[0] = ((const bf16x8*)row)[1];
    ((bf16x8*)(tb + (ln + 16) * 16))[1] = ((const bf16x8*)row)[3];
}

// ---------------------------------------------------------------- packed expert GEMM
// 64x256 per block; 8 waves (2 row-groups x 4 col-groups); per K-step each
// wave: 2 A-frags + 4 B-frags straight from global (bf16), 8 WMMAs,
// no LDS staging, no barriers in the loop.
__global__ void __launch_bounds__(256) moe_expert_gemm_packed(
    const __bf16* __restrict__ xb, const __bf16* __restrict__ wp,
    const float* __restrict__ eb, const int*    __restrict__ cnt,
    const int*   __restrict__ tok, const float* __restrict__ wgt,
    float* __restrict__ out)
{
    const int e        = blockIdx.z;
    const int count    = cnt[e];
    const int row_base = blockIdx.y * PBM;
    if (row_base >= count) return;
    const int n_base   = blockIdx.x * PBN;

    __shared__ int   s_tok[PBM];
    __shared__ float s_wgt[PBM];

    const int tid = threadIdx.x;
    if (tid < PBM) {
        int  r     = row_base + tid;
        bool valid = r < count;
        s_tok[tid] = valid ? tok[e * B_TOK + r] : -1;
        s_wgt[tid] = valid ? wgt[e * B_TOK + r] : 0.f;
    }
    __syncthreads();

    const int wid     = tid >> 5;
    const int lane    = tid & 31;
    const int waveRow = wid >> 2;          // 0..1 -> 32-row slice
    const int waveCol = wid & 3;           // 0..3 -> 64-col slice
    const int Mbase   = waveRow * 32;
    const int nW      = waveCol * 64;

    // per-lane gathered A row pointers (clamped; invalid rows get weight 0)
    const int l15 = lane & 15;
    const int kbA = (lane < 16) ? 0 : 8;
    int t0 = s_tok[Mbase + l15];      if (t0 < 0) t0 = 0;
    int t1 = s_tok[Mbase + 16 + l15]; if (t1 < 0) t1 = 0;
    const __bf16* a0p = xb + (size_t)t0 * DIM + kbA;
    const __bf16* a1p = xb + (size_t)t1 * DIM + kbA;

    // packed-W base: this wave's first col-tile, lane slice
    const __bf16* bbase = wp +
        (((size_t)e * NTILES + (n_base >> 4) + waveCol * 4) * KBLKS) * 512 +
        lane * 16;

    v8f acc[2][4] = {};

#pragma unroll 2
    for (int kb = 0; kb < KBLKS; ++kb) {
        // A fragments: two aligned 16B loads each, direct from bf16 x rows
        v16bf af0, af1;
        ((bf16x8*)&af0)[0] = *(const bf16x8*)(a0p + kb * 32);
        ((bf16x8*)&af0)[1] = *(const bf16x8*)(a0p + kb * 32 + 16);
        ((bf16x8*)&af1)[0] = *(const bf16x8*)(a1p + kb * 32);
        ((bf16x8*)&af1)[1] = *(const bf16x8*)(a1p + kb * 32 + 16);

#pragma unroll
        for (int s = 0; s < 4; ++s) {
            const __bf16* bp = bbase + ((size_t)s * KBLKS + kb) * 512;
            v16bf bfrag;
            ((bf16x8*)&bfrag)[0] = ((const bf16x8*)bp)[0];
            ((bf16x8*)&bfrag)[1] = ((const bf16x8*)bp)[1];
            acc[0][s] = __builtin_amdgcn_wmma_f32_16x16x32_bf16(
                false, af0, false, bfrag, (short)0, acc[0][s], false, false);
            acc[1][s] = __builtin_amdgcn_wmma_f32_16x16x32_bf16(
                false, af1, false, bfrag, (short)0, acc[1][s], false, false);
        }
    }

    // ---- epilogue: bias + relu + routing weight, scatter-add
    const int mHi = (lane < 16) ? 0 : 8;
#pragma unroll
    for (int g = 0; g < 2; ++g) {
#pragma unroll
        for (int s = 0; s < 4; ++s) {
            const int   col  = n_base + nW + s * 16 + (lane & 15);
            const float bias = eb[e * DIM + col];
#pragma unroll
            for (int v = 0; v < 8; ++v) {
                const int M = Mbase + g * 16 + mHi + v;
                if (row_base + M < count) {
                    float val = fmaxf(acc[g][s][v] + bias, 0.f) * s_wgt[M];
                    unsafeAtomicAdd(out + (size_t)s_tok[M] * DIM + col, val);
                }
            }
        }
    }
}

// ---------------------------------------------------------------- fallback GEMM
// (in-kernel transpose+convert; used only if workspace too small)
__global__ void __launch_bounds__(256) moe_expert_gemm_fb(
    const float* __restrict__ x,  const float* __restrict__ ew,
    const float* __restrict__ eb, const int*   __restrict__ cnt,
    const int*   __restrict__ tok, const float* __restrict__ wgt,
    float* __restrict__ out)
{
    const int e        = blockIdx.z;
    const int count    = cnt[e];
    const int row_base = blockIdx.y * BM;
    if (row_base >= count) return;
    const int n_base   = blockIdx.x * BN;

    __shared__ __bf16 Asm[BM][BK];
    __shared__ __bf16 Bsm[BN][BK];
    __shared__ int    s_tok[BM];
    __shared__ float  s_wgt[BM];

    const int tid = threadIdx.x;
    if (tid < BM) {
        int  r     = row_base + tid;
        bool valid = r < count;
        s_tok[tid] = valid ? tok[e * B_TOK + r] : -1;
        s_wgt[tid] = valid ? wgt[e * B_TOK + r] : 0.f;
    }
    __syncthreads();

    const int wid = tid >> 5, lane = tid & 31;
    const int Mbase = (wid >> 2) * 16;
    const int nW    = (wid & 3) * 64;
    const float* ewp = ew + (size_t)e * DIM * DIM;

    const int arow = tid >> 3;
    const int acol = (tid & 7) * 4;
    const int a_t  = s_tok[arow];

    v8f acc[4] = {};

    for (int k0 = 0; k0 < DIM; k0 += BK) {
        float4 av = make_float4(0.f, 0.f, 0.f, 0.f);
        if (a_t >= 0)
            av = *(const float4*)(x + (size_t)a_t * DIM + k0 + acol);
        Asm[arow][acol + 0] = f2bf(av.x);
        Asm[arow][acol + 1] = f2bf(av.y);
        Asm[arow][acol + 2] = f2bf(av.z);
        Asm[arow][acol + 3] = f2bf(av.w);
#pragma unroll
        for (int i = 0; i < 8; ++i) {
            int idx = tid + 256 * i;
            int kk  = idx >> 6;
            int nn  = (idx & 63) * 4;
            float4 wv = *(const float4*)(ewp + (size_t)(k0 + kk) * DIM + n_base + nn);
            Bsm[nn + 0][kk] = f2bf(wv.x);
            Bsm[nn + 1][kk] = f2bf(wv.y);
            Bsm[nn + 2][kk] = f2bf(wv.z);
            Bsm[nn + 3][kk] = f2bf(wv.w);
        }
        __syncthreads();

        const int mrow = Mbase + (lane & 15);
        const int kbA  = (lane < 16) ? 0 : 8;
        v16bf afrag;
        ((bf16x8*)&afrag)[0] = *(const bf16x8*)&Asm[mrow][kbA];
        ((bf16x8*)&afrag)[1] = *(const bf16x8*)&Asm[mrow][kbA + 16];

        const int kbB = (lane < 16) ? 0 : 8;
#pragma unroll
        for (int s = 0; s < 4; ++s) {
            const int ncol = nW + s * 16 + (lane & 15);
            v16bf bfrag;
            ((bf16x8*)&bfrag)[0] = *(const bf16x8*)&Bsm[ncol][kbB];
            ((bf16x8*)&bfrag)[1] = *(const bf16x8*)&Bsm[ncol][kbB + 16];
            acc[s] = __builtin_amdgcn_wmma_f32_16x16x32_bf16(
                false, afrag, false, bfrag, (short)0, acc[s], false, false);
        }
        __syncthreads();
    }

    const int mHi = (lane < 16) ? 0 : 8;
#pragma unroll
    for (int s = 0; s < 4; ++s) {
        const int   col  = n_base + nW + s * 16 + (lane & 15);
        const float bias = eb[e * DIM + col];
#pragma unroll
        for (int v = 0; v < 8; ++v) {
            const int M = Mbase + mHi + v;
            if (row_base + M < count) {
                float val = fmaxf(acc[s][v] + bias, 0.f) * s_wgt[M];
                unsafeAtomicAdd(out + (size_t)s_tok[M] * DIM + col, val);
            }
        }
    }
}

// ---------------------------------------------------------------- launcher
extern "C" void kernel_launch(void* const* d_in, const int* in_sizes, int n_in,
                              void* d_out, int out_size, void* d_ws, size_t ws_size,
                              hipStream_t stream) {
    (void)in_sizes; (void)n_in; (void)out_size;

    const float* x  = (const float*)d_in[0];   // (B, D)
    const float* ew = (const float*)d_in[1];   // (E, D, D)
    const float* eb = (const float*)d_in[2];   // (E, D)
    const float* gw = (const float*)d_in[3];   // (D, E)
    const float* gb = (const float*)d_in[4];   // (E,)
    float* out = (float*)d_out;                // (B, D)

    // ws layout: [cnt 256B][tok E*B i32][wgt E*B f32][wpack E*D*D bf16][xb B*D bf16]
    const size_t tok_off  = 256;
    const size_t wgt_off  = tok_off + sizeof(int)   * (size_t)NEXP * B_TOK;
    const size_t pack_off = (wgt_off + sizeof(float) * (size_t)NEXP * B_TOK + 255) & ~(size_t)255;
    const size_t pack_bytes = sizeof(__bf16) * (size_t)NEXP * DIM * DIM;
    const size_t xb_off   = pack_off + pack_bytes;
    const size_t xb_bytes = sizeof(__bf16) * (size_t)B_TOK * DIM;

    int*    cnt = (int*)d_ws;
    int*    tok = (int*)((char*)d_ws + tok_off);
    float*  wgt = (float*)((char*)d_ws + wgt_off);
    __bf16* wp  = (__bf16*)((char*)d_ws + pack_off);
    __bf16* xb  = (__bf16*)((char*)d_ws + xb_off);
    const bool use_pack = ws_size >= xb_off + xb_bytes;

    long n4 = ((long)B_TOK * DIM) >> 2;
    moe_zero<<<8192, 256, 0, stream>>>(out, cnt, n4);

    moe_gate<<<B_TOK / 8, 256, 0, stream>>>(x, gw, gb, cnt, tok, wgt);

    if (use_pack) {
        long n8 = ((long)B_TOK * DIM) >> 3;
        moe_pack_x<<<4096, 256, 0, stream>>>(x, xb, n8);
        dim3 pg(DIM / 256, KBLKS, NEXP);               // (8, 64, 8)
        moe_pack_w<<<pg, 256, 0, stream>>>(ew, wp);
        dim3 gg(DIM / PBN, B_TOK / PBM, NEXP);         // (8, 256, 8)
        moe_expert_gemm_packed<<<gg, 256, 0, stream>>>(xb, wp, eb, cnt, tok, wgt, out);
    } else {
        dim3 gg(DIM / BN, B_TOK / BM, NEXP);           // (8, 512, 8)
        moe_expert_gemm_fb<<<gg, 256, 0, stream>>>(x, ew, eb, cnt, tok, wgt, out);
    }
}
